// Mamba_35510789603575
// MI455X (gfx1250) — compile-verified, matrix operand order checked by reading, not run on hardware
//
#include <hip/hip_runtime.h>

// ---------------------------------------------------------------------------
// Mamba forward (8 layers, SEQ=2048, d_model=1024, d_inner=2048, n=16)
// All dense projections + tied-embedding logits run through
// v_wmma_f32_16x16x32_bf16 (f32 accumulate, on-the-fly fp32->bf16 staging).
// BK=64: 16 WMMAs per staging round per wave. Selective scan: 1 thread per
// channel, h[16] in VGPRs, B/C chunks staged via inline-asm
// global_load_async_to_lds_b128 (ASYNCcnt) + s_wait_asynccnt.
// Workspace use: ~97 MB of d_ws (fp32 activations).
// ---------------------------------------------------------------------------

#define SEQ_L   2048
#define DMODEL  1024
#define DINNER  2048
#define DSTATE  16
#define DTRANK  64
#define KCONV_  4
#define NLAYER  8
#define VOCAB_  32000

typedef __attribute__((ext_vector_type(16))) __bf16 v16bf;
typedef __attribute__((ext_vector_type(8)))  __bf16 v8bf;
typedef __attribute__((ext_vector_type(8)))  float  v8f;

// fp32 pair -> packed bf16 dword (native cvt via __bf16 fptrunc)
__device__ __forceinline__ unsigned pack2bf(float a, float b) {
  unsigned short lo = __builtin_bit_cast(unsigned short, (__bf16)a);
  unsigned short hi = __builtin_bit_cast(unsigned short, (__bf16)b);
  return (unsigned)lo | ((unsigned)hi << 16);
}

__device__ __forceinline__ float silu_f(float v) { return v / (1.0f + __expf(-v)); }

// async global->LDS 16B copy, per-lane addresses; tracked by ASYNCcnt.
// vdst = LDS byte address (low 32 bits of generic shared pointer),
// vaddr = 64-bit global address.
__device__ __forceinline__ void async_copy_b128(const void* gsrc, void* lds_dst) {
  unsigned lds_addr = (unsigned)(unsigned long long)lds_dst;
  asm volatile("global_load_async_to_lds_b128 %0, %1, off"
               :: "v"(lds_addr), "v"(gsrc)
               : "memory");
}

__device__ __forceinline__ void async_wait0() {
#if defined(__has_builtin)
#if __has_builtin(__builtin_amdgcn_s_wait_asynccnt)
  __builtin_amdgcn_s_wait_asynccnt(0);
#else
  asm volatile("s_wait_asynccnt 0x0" ::: "memory");
#endif
#else
  asm volatile("s_wait_asynccnt 0x0" ::: "memory");
#endif
}

// ---------------------------------------------------------------------------
// Generic C[M,N] = A[M,K] @ B[N,K]^T (+ addend), bf16 WMMA, f32 accumulate.
// Block tile 128x128, K-tile 64, 8 waves; each wave owns a 32x64 subtile
// (2x4 fragments of 16x16) -> 16 WMMAs per K-tile per wave.
// ---------------------------------------------------------------------------
#define BM 128
#define BN 128
#define BK 64
#define LDT 72   // padded row stride in bf16 elems (144B: keeps 16B alignment)

__global__ __launch_bounds__(256)
void mamba_gemm_bf16_wmma(const float* __restrict__ A, const float* __restrict__ B,
                          const float* __restrict__ addend, float* __restrict__ C,
                          int M, int N, int K) {
  __shared__ __bf16 sA[BM * LDT];
  __shared__ __bf16 sB[BN * LDT];

  const int tid   = threadIdx.x;
  const int lane  = tid & 31;
  const int wave  = tid >> 5;
  const int wm    = (wave & 3) * 32;   // wave's M offset inside tile
  const int wn    = (wave >> 2) * 64;  // wave's N offset inside tile
  const int tileM = blockIdx.y * BM;
  const int tileN = blockIdx.x * BN;

  const int row16 = lane & 15;         // fragment row (M or N)
  const int khalf = (lane >> 4) << 3;  // 0 (lanes 0-15) or 8 (lanes 16-31)

  v8f zero = {};
  v8f acc[2][4];
#pragma unroll
  for (int i = 0; i < 2; ++i)
#pragma unroll
    for (int j = 0; j < 4; ++j) acc[i][j] = zero;

  for (int k0 = 0; k0 < K; k0 += BK) {
    // ---- stage A/B tiles: 128x64 fp32 each, converted to bf16 -------------
#pragma unroll
    for (int it = 0; it < 8; ++it) {
      int idx = tid + it * 256;        // float4 slot 0..2047
      int r   = idx >> 4;              // tile row 0..127
      int kc  = (idx & 15) << 2;       // k offset 0,4,...,60
      float4 va = make_float4(0.f, 0.f, 0.f, 0.f);
      float4 vb = make_float4(0.f, 0.f, 0.f, 0.f);
      int ga = tileM + r, gb = tileN + r;
      if (ga < M) va = *(const float4*)(A + (size_t)ga * K + k0 + kc);
      if (gb < N) vb = *(const float4*)(B + (size_t)gb * K + k0 + kc);
      unsigned* da = (unsigned*)&sA[r * LDT + kc];
      unsigned* db = (unsigned*)&sB[r * LDT + kc];
      da[0] = pack2bf(va.x, va.y);  da[1] = pack2bf(va.z, va.w);
      db[0] = pack2bf(vb.x, vb.y);  db[1] = pack2bf(vb.z, vb.w);
    }
    __syncthreads();

    // prefetch next K-tile into L2 while we compute (global_prefetch_b8)
    if (k0 + BK < K) {
      int r = tid & 127;
      if (tid < 128) {
        int ga = tileM + r; if (ga >= M) ga = M - 1;
        __builtin_prefetch(A + (size_t)ga * K + k0 + BK, 0, 1);
      } else {
        int gb = tileN + r; if (gb >= N) gb = N - 1;
        __builtin_prefetch(B + (size_t)gb * K + k0 + BK, 0, 1);
      }
    }

    // ---- two K=32 WMMA steps per staged tile ------------------------------
#pragma unroll
    for (int ks = 0; ks < 2; ++ks) {
      const int kb = ks * 32;
      v16bf afrag[2], bfrag[4];
#pragma unroll
      for (int i = 0; i < 2; ++i) {
        const __bf16* p = &sA[(wm + i * 16 + row16) * LDT + kb + khalf];
        v8bf lo = *(const v8bf*)p;         // K = kb+khalf .. +7
        v8bf hi = *(const v8bf*)(p + 16);  // K = kb+16+khalf .. +7
#pragma unroll
        for (int e = 0; e < 8; ++e) { afrag[i][e] = lo[e]; afrag[i][8 + e] = hi[e]; }
      }
#pragma unroll
      for (int j = 0; j < 4; ++j) {
        const __bf16* p = &sB[(wn + j * 16 + row16) * LDT + kb + khalf];
        v8bf lo = *(const v8bf*)p;
        v8bf hi = *(const v8bf*)(p + 16);
#pragma unroll
        for (int e = 0; e < 8; ++e) { bfrag[j][e] = lo[e]; bfrag[j][8 + e] = hi[e]; }
      }
#pragma unroll
      for (int i = 0; i < 2; ++i)
#pragma unroll
        for (int j = 0; j < 4; ++j)
          acc[i][j] = __builtin_amdgcn_wmma_f32_16x16x32_bf16(
              false, afrag[i], false, bfrag[j], (short)0, acc[i][j], false, false);
    }
    __syncthreads();
  }

  // ---- epilogue: C/D layout -> VGPR r holds M=r (+8 for lanes 16-31) ------
  const int mofs = (lane >> 4) * 8;
  const int nlan = lane & 15;
#pragma unroll
  for (int i = 0; i < 2; ++i) {
#pragma unroll
    for (int j = 0; j < 4; ++j) {
      int ng = tileN + wn + j * 16 + nlan;
      if (ng >= N) continue;
#pragma unroll
      for (int r = 0; r < 8; ++r) {
        int mg = tileM + wm + i * 16 + mofs + r;
        if (mg >= M) continue;
        size_t cidx = (size_t)mg * N + ng;
        float v = acc[i][j][r];
        if (addend) v += addend[cidx];
        C[cidx] = v;
      }
    }
  }
}

// ---------------------------------------------------------------------------
// Embedding gather: x[l,:] = embed[ids[l],:]
// ---------------------------------------------------------------------------
__global__ void mamba_gather(const int* __restrict__ ids,
                             const float* __restrict__ embed,
                             float* __restrict__ x) {
  int l = blockIdx.x;
  int id = ids[l];
  const float4* src = (const float4*)(embed + (size_t)id * DMODEL);
  float4* dst = (float4*)(x + (size_t)l * DMODEL);
  for (int c = threadIdx.x; c < DMODEL / 4; c += blockDim.x) dst[c] = src[c];
}

// ---------------------------------------------------------------------------
// RMS pre-norm (no learned scale): one block per row
// ---------------------------------------------------------------------------
__global__ __launch_bounds__(256)
void mamba_rmsnorm(const float* __restrict__ x, float* __restrict__ xn) {
  __shared__ float red[8];
  __shared__ float scale;
  int l = blockIdx.x;
  const float* row = x + (size_t)l * DMODEL;
  float s = 0.f;
  for (int c = threadIdx.x; c < DMODEL; c += 256) { float v = row[c]; s += v * v; }
#pragma unroll
  for (int off = 16; off > 0; off >>= 1) s += __shfl_down(s, off, 32);
  if ((threadIdx.x & 31) == 0) red[threadIdx.x >> 5] = s;
  __syncthreads();
  if (threadIdx.x == 0) {
    float t = 0.f;
#pragma unroll
    for (int i = 0; i < 8; ++i) t += red[i];
    scale = rsqrtf(t / (float)DMODEL + 1e-6f);
  }
  __syncthreads();
  float sc = scale;
  for (int c = threadIdx.x; c < DMODEL; c += 256)
    xn[(size_t)l * DMODEL + c] = row[c] * sc;
}

// ---------------------------------------------------------------------------
// Causal depthwise conv (K=4) + SiLU
// ---------------------------------------------------------------------------
__global__ __launch_bounds__(256)
void mamba_conv_silu(const float* __restrict__ u, const float* __restrict__ w,
                     float* __restrict__ out) {
  int idx = blockIdx.x * 256 + threadIdx.x;
  if (idx >= SEQ_L * DINNER) return;
  int d = idx % DINNER;
  int l = idx / DINNER;
  float acc = 0.f;
#pragma unroll
  for (int k = 0; k < KCONV_; ++k) {
    int li = l - (KCONV_ - 1) + k;
    if (li >= 0) acc += w[d * KCONV_ + k] * u[(size_t)li * DINNER + d];
  }
  out[idx] = silu_f(acc);
}

// ---------------------------------------------------------------------------
// dt = softplus(dt_raw + bias)
// ---------------------------------------------------------------------------
__global__ __launch_bounds__(256)
void mamba_softplus_bias(float* __restrict__ dt, const float* __restrict__ b) {
  int idx = blockIdx.x * 256 + threadIdx.x;
  if (idx >= SEQ_L * DINNER) return;
  float v = dt[idx] + b[idx % DINNER];
  dt[idx] = (v > 20.f) ? v : log1pf(__expf(v));
}

// ---------------------------------------------------------------------------
// Selective scan (bilinear discretization), fused with +u*D and silu(res) gate.
// One thread per channel; h[16] in VGPRs; B/C staged in 128-step LDS chunks
// via global_load_async_to_lds_b128 (ASYNCcnt).
// ---------------------------------------------------------------------------
#define SCHUNK 128
__global__ __launch_bounds__(256)
void mamba_scan(const float* __restrict__ u, const float* __restrict__ dt,
                const float* __restrict__ logA, const float* __restrict__ Bm,
                const float* __restrict__ Cm, const float* __restrict__ Dp,
                const float* __restrict__ res, float* __restrict__ y) {
  __shared__ float sB[SCHUNK * DSTATE];
  __shared__ float sC[SCHUNK * DSTATE];
  const int d = blockIdx.x * 256 + threadIdx.x;

  float Areg[DSTATE], h[DSTATE];
#pragma unroll
  for (int n = 0; n < DSTATE; ++n) {
    Areg[n] = -__expf(logA[(size_t)d * DSTATE + n]);
    h[n] = 0.f;
  }
  const float dp = Dp[d];

  for (int c0 = 0; c0 < SEQ_L; c0 += SCHUNK) {
    __syncthreads();
    // async global->LDS copy: 8KB per array, 16B per lane-op, 2 ops/thread
#pragma unroll
    for (int it = 0; it < 2; ++it) {
      int fo = (threadIdx.x + it * 256) * 4;  // float offset, 16B chunks
      async_copy_b128(Bm + (size_t)c0 * DSTATE + fo, sB + fo);
      async_copy_b128(Cm + (size_t)c0 * DSTATE + fo, sC + fo);
    }
    async_wait0();
    __syncthreads();
    for (int t = 0; t < SCHUNK; ++t) {
      int l = c0 + t;
      float ul  = u[(size_t)l * DINNER + d];
      float dtl = dt[(size_t)l * DINNER + d];
      float yl = 0.f;
#pragma unroll
      for (int n = 0; n < DSTATE; ++n) {
        float dtA2 = 0.5f * dtl * Areg[n];
        float r = 1.0f / (1.0f - dtA2);        // 1/denom
        float dA = (1.0f + dtA2) * r;
        float dB = dtl * sB[t * DSTATE + n] * r;
        h[n] = dA * h[n] + dB * ul;
        yl += h[n] * sC[t * DSTATE + n];
      }
      float rv = res[(size_t)l * DINNER + d];
      y[(size_t)l * DINNER + d] = (yl + ul * dp) * silu_f(rv);
    }
  }
}

// ---------------------------------------------------------------------------
// Host orchestration
// ---------------------------------------------------------------------------
extern "C" void kernel_launch(void* const* d_in, const int* in_sizes, int n_in,
                              void* d_out, int out_size, void* d_ws, size_t ws_size,
                              hipStream_t stream) {
  const int*   ids   = (const int*)  d_in[0];
  const float* embed = (const float*)d_in[1];   // [VOCAB, DMODEL]
  const float* Wres  = (const float*)d_in[2];   // [L, DINNER, DMODEL]
  const float* Win   = (const float*)d_in[3];   // [L, DINNER, DMODEL]
  const float* Wconv = (const float*)d_in[4];   // [L, DINNER, 1, K]
  const float* WB    = (const float*)d_in[5];   // [L, DSTATE, DINNER]
  const float* WC    = (const float*)d_in[6];   // [L, DSTATE, DINNER]
  const float* Wdt1  = (const float*)d_in[7];   // [L, DTRANK, DINNER]
  const float* Wdt2  = (const float*)d_in[8];   // [L, DINNER, DTRANK]
  const float* bdt   = (const float*)d_in[9];   // [L, DINNER]
  const float* logA  = (const float*)d_in[10];  // [L, DINNER, DSTATE]
  const float* Dp    = (const float*)d_in[11];  // [L, DINNER]
  const float* Wout  = (const float*)d_in[12];  // [L, DMODEL, DINNER]
  (void)in_sizes; (void)n_in; (void)out_size; (void)ws_size;

  float* ws  = (float*)d_ws;
  float* x   = ws;                               // [SEQ, DMODEL]
  float* xn  = x   + (size_t)SEQ_L * DMODEL;     // [SEQ, DMODEL]
  float* res = xn  + (size_t)SEQ_L * DMODEL;     // [SEQ, DINNER]
  float* u   = res + (size_t)SEQ_L * DINNER;     // [SEQ, DINNER]
  float* uc  = u   + (size_t)SEQ_L * DINNER;     // [SEQ, DINNER]
  float* Bm  = uc  + (size_t)SEQ_L * DINNER;     // [SEQ, DSTATE]
  float* Cm  = Bm  + (size_t)SEQ_L * DSTATE;     // [SEQ, DSTATE]
  float* dtl = Cm  + (size_t)SEQ_L * DSTATE;     // [SEQ, DTRANK]
  float* dtf = dtl + (size_t)SEQ_L * DTRANK;     // [SEQ, DINNER]
  float* y   = dtf + (size_t)SEQ_L * DINNER;     // [SEQ, DINNER]

  auto GEMM = [&](const float* A, const float* B, const float* add, float* C,
                  int M, int N, int K) {
    dim3 grid((N + BN - 1) / BN, (M + BM - 1) / BM);
    mamba_gemm_bf16_wmma<<<grid, 256, 0, stream>>>(A, B, add, C, M, N, K);
  };

  mamba_gather<<<SEQ_L, 256, 0, stream>>>(ids, embed, x);

  for (int i = 0; i < NLAYER; ++i) {
    mamba_rmsnorm<<<SEQ_L, 256, 0, stream>>>(x, xn);
    GEMM(xn, Wres + (size_t)i * DINNER * DMODEL, nullptr, res, SEQ_L, DINNER, DMODEL);
    GEMM(xn, Win  + (size_t)i * DINNER * DMODEL, nullptr, u,   SEQ_L, DINNER, DMODEL);
    mamba_conv_silu<<<(SEQ_L * DINNER + 255) / 256, 256, 0, stream>>>(
        u, Wconv + (size_t)i * DINNER * KCONV_, uc);
    GEMM(uc, WB   + (size_t)i * DSTATE * DINNER, nullptr, Bm,  SEQ_L, DSTATE, DINNER);
    GEMM(uc, WC   + (size_t)i * DSTATE * DINNER, nullptr, Cm,  SEQ_L, DSTATE, DINNER);
    GEMM(uc, Wdt1 + (size_t)i * DTRANK * DINNER, nullptr, dtl, SEQ_L, DTRANK, DINNER);
    GEMM(dtl, Wdt2 + (size_t)i * DINNER * DTRANK, nullptr, dtf, SEQ_L, DINNER, DTRANK);
    mamba_softplus_bias<<<(SEQ_L * DINNER + 255) / 256, 256, 0, stream>>>(
        dtf, bdt + (size_t)i * DINNER);
    mamba_scan<<<DINNER / 256, 256, 0, stream>>>(
        uc, dtf, logA + (size_t)i * DINNER * DSTATE, Bm, Cm,
        Dp + (size_t)i * DINNER, res, y);
    // x = x + y @ Wout^T  (in-place residual add via epilogue)
    GEMM(y, Wout + (size_t)i * DMODEL * DINNER, x, x, SEQ_L, DMODEL, DINNER);
  }

  // logits = x @ embed^T  (134 GFLOP; embed fits in 192 MB L2)
  GEMM(x, embed, nullptr, (float*)d_out, SEQ_L, VOCAB_, DMODEL);
}
